// MoeBlock_52793738003150
// MI455X (gfx1250) — matrix operand order repaired and checked by reading, not compiled, not run on hardware
//
#include <hip/hip_runtime.h>

typedef __attribute__((ext_vector_type(16))) _Float16 v16h;
typedef __attribute__((ext_vector_type(8)))  _Float16 v8h;
typedef __attribute__((ext_vector_type(8)))  float    v8f;

#define B_    32
#define CIN   256
#define COUT  256
#define H_    56
#define W_    56
#define HP    58              // padded spatial
#define HW    (H_*W_)         // 3136
#define NPIX  (B_*HW)         // 100352
#define NCHNK 72              // (9*256)/32 K-chunks

// ---------------- kernel 1: gate-weighted expert bias (32x256 GEMM) --------
__global__ void bias_kernel(const float* __restrict__ gate,
                            const float* __restrict__ bexp,
                            float* __restrict__ bias) {
    int t = blockIdx.x * 256 + threadIdx.x;     // 8192 total
    int b = t >> 8, co = t & 255;
    float s = 0.f;
    for (int i = 0; i < CIN; ++i) s += gate[b * CIN + i] * bexp[i * COUT + co];
    bias[t] = s;
}

// ---------------- kernel 2: pack weights fp32 OIHW -> f16 [e][co][ci] ------
__global__ void pack_w(const float* __restrict__ W, _Float16* __restrict__ Whp) {
    int t = blockIdx.x * 256 + threadIdx.x;     // 9*256*256 = 589824
    int ci = t & 255;
    int co = (t >> 8) & 255;
    int e  = t >> 16;                           // 0..8 = ky*3+kx
    Whp[t] = (_Float16)W[(co * CIN + ci) * 9 + e];
}

// ---------------- kernel 3: x*gate, NCHW fp32 -> padded NHWC f16 -----------
__global__ void pack_x(const float* __restrict__ x, const float* __restrict__ gate,
                       _Float16* __restrict__ xh) {
    __shared__ float tile[32][33];
    int ci0 = blockIdx.x * 32;
    int w0  = blockIdx.y * 32;
    int bh  = blockIdx.z;
    int b = bh / H_, h = bh % H_;
    int t = threadIdx.x;
    int tw = t & 31, ti = t >> 5;               // ti = 0..7
#pragma unroll
    for (int rr = 0; rr < 4; ++rr) {
        int cil = ti + rr * 8;
        int ci  = ci0 + cil;
        int w   = w0 + tw;
        if (w < W_)
            tile[cil][tw] = x[((long)(b * CIN + ci) * H_ + h) * W_ + w] * gate[b * CIN + ci];
    }
    __syncthreads();
#pragma unroll
    for (int rr = 0; rr < 4; ++rr) {
        int wl = ti + rr * 8;
        int w  = w0 + wl;
        int ci = ci0 + tw;
        if (w < W_)
            xh[(((long)b * HP + (h + 1)) * HP + (w + 1)) * CIN + ci] = (_Float16)tile[tw][wl];
    }
}

// fragment load: 16 contiguous halves from LDS (two b128 loads, no VALU shuffle)
__device__ inline v16h ld_frag(const _Float16* p) {
    v8h lo = *(const v8h*)p;
    v8h hi = *(const v8h*)(p + 8);
    return __builtin_shufflevector(lo, hi, 0, 1, 2, 3, 4, 5, 6, 7,
                                   8, 9, 10, 11, 12, 13, 14, 15);
}

__device__ inline v8f wmma16(v16h a, v16h b, v8f c) {
    return __builtin_amdgcn_wmma_f32_16x16x32_f16(false, a, false, b,
                                                  (short)0, c, false, false);
}

// low 32 bits of a generic LDS pointer == wave-relative LDS byte address
__device__ inline unsigned lds_addr(const void* p) {
    return (unsigned)(unsigned long long)p;
}

// ---------------- kernel 4: implicit-GEMM conv via WMMA --------------------
// A = Whp[e][co][ci] (M=Cout x K), B = padded NHWC patches (K x N=pixels)
// block tile: 64 Cout x 128 pixels; 8 waves as 2(M) x 4(N), each wave a
// 32x32 macro-tile = 2x2 WMMA tiles. Global->LDS staging uses the CDNA5
// async-to-LDS path (ASYNCcnt) with double-buffered LDS.
__global__ __launch_bounds__(256)
void conv_wmma(const _Float16* __restrict__ xh, const _Float16* __restrict__ Whp,
               const float* __restrict__ bias, float* __restrict__ y) {
    __shared__ _Float16 As[2][64][40];    // 2 x 5120 B, row stride 80 B
    __shared__ _Float16 Bs[2][128][40];   // 2 x 10240 B

    const int t    = threadIdx.x;
    const int lane = t & 31;
    const int wave = t >> 5;
    const int co0   = blockIdx.y * 64;
    const int pbase = blockIdx.x * 128;

    // --- cooperative loader geometry ---
    const int arow = t >> 2, aseg = t & 3;
    const long aGbase = (long)(co0 + arow) * CIN + aseg * 8;
    // A stored permuted (seg order 0,2,1,3) so lane fragments are contiguous:
    // halves [0..15] = K{0..7,16..23}, halves [16..31] = K{8..15,24..31}
    const int aoff = (aseg & 1) * 16 + (aseg >> 1) * 8;
    const int brow0 = t >> 2, brow1 = brow0 + 64, bseg = t & 3;
    const int p0 = pbase + brow0;
    const int p1 = pbase + brow1;
    const int pb0 = p0 / HW, ph0 = (p0 % HW) / W_, pw0 = p0 % W_;
    const int pb1 = p1 / HW, ph1 = (p1 % HW) / W_, pw1 = p1 % W_;
    const long bGbase0 = (((long)pb0 * HP + ph0) * HP + pw0) * CIN + bseg * 8;
    const long bGbase1 = (((long)pb1 * HP + ph1) * HP + pw1) * CIN + bseg * 8;

    // issue the 3 async global->LDS b128 copies for K-chunk kn into buffer buf
    auto issue = [&](int kn, int buf) {
        int e   = kn >> 3;                 // tap 0..8
        int ci0 = (kn & 7) * 32;
        int ky = e / 3, kx = e - ky * 3;
        long koff = (long)(ky * HP + kx) * CIN + ci0;
        const _Float16* ga  = Whp + aGbase + (long)e * CIN * COUT + ci0;
        const _Float16* gb0 = xh + bGbase0 + koff;
        const _Float16* gb1 = xh + bGbase1 + koff;
        unsigned la  = lds_addr(&As[buf][arow][aoff]);
        unsigned lb0 = lds_addr(&Bs[buf][brow0][(bseg & 1) * 16 + (bseg >> 1) * 8]);
        unsigned lb1 = lds_addr(&Bs[buf][brow1][(bseg & 1) * 16 + (bseg >> 1) * 8]);
        asm volatile(
            "global_load_async_to_lds_b128 %0, %1, off\n\t"
            "global_load_async_to_lds_b128 %2, %3, off\n\t"
            "global_load_async_to_lds_b128 %4, %5, off"
            :
            : "v"(la), "v"(ga), "v"(lb0), "v"(gb0), "v"(lb1), "v"(gb1)
            : "memory");
    };

    v8f acc[2][2] = {};
    const int m0   = (wave >> 2) * 32;          // 0 / 32
    const int n0   = (wave & 3) * 32;           // 0 / 32 / 64 / 96
    const int mrow = lane & 15;
    const int fo   = (lane & 16) ? 16 : 0;      // fragment half-offset selector

    issue(0, 0);                                // prologue: chunk 0 -> buf 0

    for (int kk = 0; kk < NCHNK; ++kk) {
        const int buf = kk & 1;
        if (kk + 1 < NCHNK) {
            issue(kk + 1, buf ^ 1);             // prefetch next chunk
            // drain this wave's chunk-kk copies (async loads complete in order;
            // 3 newer ones for chunk kk+1 may remain in flight)
            asm volatile("s_wait_asynccnt 0x3" ::: "memory");
        } else {
            asm volatile("s_wait_asynccnt 0x0" ::: "memory");
        }
        __syncthreads();                        // all waves' tiles resident

        v16h a0 = ld_frag(&As[buf][m0 + mrow][fo]);
        v16h a1 = ld_frag(&As[buf][m0 + 16 + mrow][fo]);
        v16h b0 = ld_frag(&Bs[buf][n0 + mrow][fo]);
        v16h b1 = ld_frag(&Bs[buf][n0 + 16 + mrow][fo]);
        acc[0][0] = wmma16(a0, b0, acc[0][0]);
        acc[0][1] = wmma16(a0, b1, acc[0][1]);
        acc[1][0] = wmma16(a1, b0, acc[1][0]);
        acc[1][1] = wmma16(a1, b1, acc[1][1]);

        __syncthreads();                        // buf free for reuse at kk+2
    }

    // epilogue: +bias, store y in final NCHW layout (into d_out)
    const int col = lane & 15;
    const int hs  = (lane >> 4) & 1;            // VGPR r -> row r / r+8
#pragma unroll
    for (int sn = 0; sn < 2; ++sn) {
        const int pg  = pbase + n0 + sn * 16 + col;  // 16-pixel groups never straddle b
        const int ob  = pg / HW;
        const int ohw = pg - ob * HW;
#pragma unroll
        for (int sm = 0; sm < 2; ++sm) {
#pragma unroll
            for (int r = 0; r < 8; ++r) {
                int co = co0 + m0 + sm * 16 + r + 8 * hs;
                float v = acc[sm][sn][r] + bias[ob * COUT + co];
                y[(long)(ob * COUT + co) * HW + ohw] = v;
            }
        }
    }
}

// ---------------- kernel 5: per-channel batch stats ------------------------
__global__ void stats_kernel(const float* __restrict__ y,
                             float* __restrict__ meanv, float* __restrict__ varv) {
    int co = blockIdx.x, t = threadIdx.x;
    float s = 0.f, s2 = 0.f;
    for (int b = 0; b < B_; ++b) {
        const float4* pp = (const float4*)(y + (long)(b * COUT + co) * HW);
        for (int i = t; i < HW / 4; i += 256) {
            float4 v = pp[i];
            s  += v.x + v.y + v.z + v.w;
            s2 += v.x * v.x + v.y * v.y + v.z * v.z + v.w * v.w;
        }
    }
    __shared__ float ls[256], ls2[256];
    ls[t] = s; ls2[t] = s2;
    __syncthreads();
    for (int o = 128; o > 0; o >>= 1) {
        if (t < o) { ls[t] += ls[t + o]; ls2[t] += ls2[t + o]; }
        __syncthreads();
    }
    if (t == 0) {
        float m = ls[0] / (float)NPIX;
        meanv[co] = m;
        varv[co]  = ls2[0] / (float)NPIX - m * m;
    }
}

// ---------------- kernel 6: fused BN + ReLU, in place on d_out -------------
__global__ void bn_relu(const float* __restrict__ meanv, const float* __restrict__ varv,
                        const float* __restrict__ gamma, const float* __restrict__ beta,
                        float* __restrict__ out) {
    long i4 = (long)blockIdx.x * 256 + threadIdx.x;     // float4 index
    long i  = i4 * 4;
    int co = (int)((i / HW) % COUT);                    // HW % 4 == 0: no straddle
    float inv = rsqrtf(varv[co] + 1e-5f);
    float sc  = inv * gamma[co];
    float sh  = beta[co] - meanv[co] * sc;
    float4* pp = (float4*)out + i4;
    float4 v = *pp;
    v.x = fmaxf(v.x * sc + sh, 0.f);
    v.y = fmaxf(v.y * sc + sh, 0.f);
    v.z = fmaxf(v.z * sc + sh, 0.f);
    v.w = fmaxf(v.w * sc + sh, 0.f);
    *pp = v;
}

// ---------------------------------------------------------------------------
extern "C" void kernel_launch(void* const* d_in, const int* in_sizes, int n_in,
                              void* d_out, int out_size, void* d_ws, size_t ws_size,
                              hipStream_t stream) {
    const float* x     = (const float*)d_in[0];   // (32,256,56,56)
    const float* gate  = (const float*)d_in[1];   // (32,256)
    const float* W     = (const float*)d_in[2];   // (256,256,3,3)
    const float* bexp  = (const float*)d_in[3];   // (256,256)
    const float* gamma = (const float*)d_in[4];   // (256,)
    const float* beta  = (const float*)d_in[5];   // (256,)
    float* out = (float*)d_out;                   // (32,256,56,56), also holds y

    const size_t XH_BYTES = (size_t)B_ * HP * HP * CIN * sizeof(_Float16); // 55,115,776
    const size_t WH_BYTES = (size_t)9 * COUT * CIN * sizeof(_Float16);     //  1,179,648
    const size_t BI_BYTES = (size_t)B_ * COUT * sizeof(float);             //     32,768
    char* ws = (char*)d_ws;
    _Float16* xh  = (_Float16*)ws;
    _Float16* Whp = (_Float16*)(ws + XH_BYTES);
    float* bias   = (float*)(ws + XH_BYTES + WH_BYTES);
    float* meanv  = (float*)(ws + XH_BYTES + WH_BYTES + BI_BYTES);
    float* varv   = meanv + COUT;

    hipMemsetAsync(xh, 0, XH_BYTES, stream);      // zero padding halo

    pack_w<<<(9 * COUT * CIN) / 256, 256, 0, stream>>>(W, Whp);
    pack_x<<<dim3(CIN / 32, 2, B_ * H_), 256, 0, stream>>>(x, gate, xh);
    bias_kernel<<<(B_ * COUT) / 256, 256, 0, stream>>>(gate, bexp, bias);

    conv_wmma<<<dim3(NPIX / 128, COUT / 64), 256, 0, stream>>>(xh, Whp, bias, out);

    stats_kernel<<<COUT, 256, 0, stream>>>(out, meanv, varv);
    bn_relu<<<(int)(((long)NPIX * COUT / 4) / 256), 256, 0, stream>>>(
        meanv, varv, gamma, beta, out);
}
